// BidirectionalAttention_49795850830307
// MI455X (gfx1250) — matrix operand order repaired and checked by reading, not compiled, not run on hardware
//
#include <hip/hip_runtime.h>

// ---------------------------------------------------------------------------
// Fused 2D-RoPE bidirectional attention (no softmax) for gfx1250 (MI455X).
//   O[b,h] = (rope(Q[b,h]) * rope(K[b,h])^T / sqrt(N)) * V[b]
// B=16 H=4 T=256 N=2048 D=256.
// Block = (b, h, 64-row M quarter), 512 threads = 16 waves.
// Per 32-k chunk: cooperatively stage rope'd bf16 Q/K chunks in LDS
// (float4 quads, double-buffered, 32-bit SADDR-form addressing), then
// bf16 WMMA into per-wave S tiles; finish with f32 16x16x4 WMMA for S*V.
// ---------------------------------------------------------------------------

typedef __attribute__((ext_vector_type(16))) __bf16 v16bf;
typedef __attribute__((ext_vector_type(8)))  float  v8f;
typedef __attribute__((ext_vector_type(2)))  float  v2f;
#if defined(__has_builtin)
#if __has_builtin(__builtin_amdgcn_cvt_pk_bf16_f32)
#define HAVE_CVT_PK_BF16 1
typedef __attribute__((ext_vector_type(2))) __bf16 v2bf;
#endif
#endif

constexpr int Bsz = 16, Hn = 4, Tn = 256, Nn = 2048, Dn = 256;

constexpr int TAB_STRIDE = 516;                 // 512 + 4 pad (uints)
constexpr int TAB_U      = 16 * TAB_STRIDE;     // 8256 uints
constexpr int ROW_U      = 20;                  // 16 bf16-pairs + 4 pad -> b128 ok
constexpr int KCH_U      = 256 * ROW_U;         // K chunk: 256 rows x 32 k
constexpr int QCH_U      = 64 * ROW_U;          // Q chunk:  64 rows x 32 k
constexpr int CH_U       = KCH_U + QCH_U;       // 6400 uints per buffer
constexpr int SMEM_U     = TAB_U + 2 * CH_U;    // 21056 uints = 84224 B
constexpr int S_STRIDE   = 260;                 // floats, 64x260 aliases smem

__device__ __forceinline__ float bf_lo(unsigned u) {
  return __builtin_bit_cast(float, (u & 0xFFFFu) << 16);
}
__device__ __forceinline__ float bf_hi(unsigned u) {
  return __builtin_bit_cast(float, u & 0xFFFF0000u);
}
__device__ __forceinline__ unsigned short f2bf(float f) {
  unsigned u = __builtin_bit_cast(unsigned, f);
  u += 0x7FFFu + ((u >> 16) & 1u);
  return (unsigned short)(u >> 16);
}
__device__ __forceinline__ unsigned pkbf(float e0, float e1) {
#ifdef HAVE_CVT_PK_BF16
  v2bf p = __builtin_amdgcn_cvt_pk_bf16_f32(e0, e1);
  return __builtin_bit_cast(unsigned, p);
#else
  return (unsigned)f2bf(e0) | ((unsigned)f2bf(e1) << 16);
#endif
}

__global__ __launch_bounds__(512) void rope_attn_gfx1250(
    const float* __restrict__ Qg, const float* __restrict__ Kg,
    const float* __restrict__ Vg, float* __restrict__ Og) {
  __shared__ __align__(16) unsigned smem[SMEM_U];
  unsigned* tab  = smem;             // [16][516] packed (cos,sin) bf16
  unsigned* bufA = smem + TAB_U;     // ping-pong chunk buffers
  unsigned* bufB = bufA + CH_U;

  const int tid  = threadIdx.x;
  const int wv   = tid >> 5;         // wave 0..15
  const int lane = tid & 31;
  const int col  = lane & 15;
  const int hh   = lane >> 4;

  const int mq    = blockIdx.x;      // 4 quarters of 64 M rows
  const int hi    = blockIdx.y;
  const int bi    = blockIdx.z;
  const int mbase = mq * 64;

  // ---- 1. angle table: ang(p,m) = 2*pi*frac(p * 10000^(-m/512))
  for (int j = 0; j < 16; ++j) {
    int e = tid + j * 512;           // 8192 entries
    int p = e >> 9, m = e & 511;
    float invf = __expf((float)m * (-9.210340371976184f / 512.0f));
    float x = (float)p * invf;
    float ang = 6.283185307179586f * (x - floorf(x));
    tab[p * TAB_STRIDE + m] = (unsigned)f2bf(__cosf(ang)) |
                              ((unsigned)f2bf(__sinf(ang)) << 16);
  }
  __syncthreads();

  // Uniform 64-bit bases; all per-thread offsets below are 32-bit (SADDR form).
  const float* Qbh = Qg + (size_t)((bi * Hn + hi) * Tn + mbase) * Nn;
  const float* Kbh = Kg + (size_t)(bi * Hn + hi) * Tn * Nn;

  // ---- rope 4 consecutive elements (one float4) using one b128 table read
  auto rope4 = [&](float4 x, int p, int m, uint2& out) {
    uint4 cs = *(const uint4*)(tab + p * TAB_STRIDE + m);   // m % 4 == 0
    float e0 = x.x * bf_lo(cs.x) - x.y * bf_hi(cs.x);
    float e1 = x.y * bf_lo(cs.y) + x.x * bf_hi(cs.y);
    float e2 = x.z * bf_lo(cs.z) - x.w * bf_hi(cs.z);
    float e3 = x.w * bf_lo(cs.w) + x.z * bf_hi(cs.w);
    out.x = pkbf(e0, e1);
    out.y = pkbf(e2, e3);
  };

  // ---- stage one 32-wide k chunk of rope'd bf16 Q(64 rows) + K(256 rows)
  auto stage = [&](int kc, unsigned* buf) {
    const int kb = kc * 32;
    {                                 // Q: 512 quads, 1 round
      int row = tid >> 3, qc = tid & 7;
      int t = mbase + row;
      int n = kb + 4 * qc;
      int p = ((n >> 9) & 1) ? (t & 15) : (t >> 4);
      float4 x = *(const float4*)(Qbh + (row * Nn + n));
      uint2 o;
      rope4(x, p, n & 511, o);
      *(uint2*)(buf + KCH_U + row * ROW_U + 2 * qc) = o;
    }
    #pragma unroll
    for (int j = 0; j < 4; ++j) {     // K: 2048 quads, 4 rounds
      int i = tid + j * 512;
      int srow = i >> 3, qc = i & 7;
      int n = kb + 4 * qc;
      int p = ((n >> 9) & 1) ? (srow & 15) : (srow >> 4);
      float4 x = *(const float4*)(Kbh + (srow * Nn + n));
      uint2 o;
      rope4(x, p, n & 511, o);
      *(uint2*)(buf + srow * ROW_U + 2 * qc) = o;
    }
  };

  // ---- 2. GEMM1: wave = (mtile mt, stile group sg), 4 S-tiles in regs
  const int mt = wv >> 2, sg = wv & 3;
  v8f acc[4] = {{0,0,0,0,0,0,0,0}, {0,0,0,0,0,0,0,0},
                {0,0,0,0,0,0,0,0}, {0,0,0,0,0,0,0,0}};

  auto compute = [&](const unsigned* buf) {
    const unsigned* qr = buf + KCH_U + (mt * 16 + col) * ROW_U;
    union { v16bf v; uint4 q[2]; } af;
    af.q[0] = *(const uint4*)(qr + 4 * hh);       // K elems [8hh, 8hh+8)
    af.q[1] = *(const uint4*)(qr + 8 + 4 * hh);   // K elems [16+8hh, ..)
    #pragma unroll
    for (int si = 0; si < 4; ++si) {
      const unsigned* kr = buf + ((sg * 4 + si) * 16 + col) * ROW_U + 4 * hh;
      union { v16bf v; uint4 q[2]; } bf;
      bf.q[0] = *(const uint4*)(kr);
      bf.q[1] = *(const uint4*)(kr + 8);
      acc[si] = __builtin_amdgcn_wmma_f32_16x16x32_bf16(
          false, af.v, false, bf.v, (short)0, acc[si], false, false);
    }
  };

  stage(0, bufA);
  __syncthreads();
  for (int kc = 0; kc < 64; ++kc) {
    unsigned* cur = (kc & 1) ? bufB : bufA;
    unsigned* nxt = (kc & 1) ? bufA : bufB;
    if (kc < 63) stage(kc + 1, nxt);   // overlap next-chunk staging
    compute(cur);
    __syncthreads();                   // one barrier per chunk (ping-pong)
  }

  // ---- 3. spill scaled S (64 x 256) to LDS (aliases table + buffers)
  float* Sl = (float*)smem;
  const float scale = 0.022097086912079608f;     // 1/sqrt(2048)
  #pragma unroll
  for (int si = 0; si < 4; ++si)
    #pragma unroll
    for (int r = 0; r < 8; ++r)
      Sl[(mt * 16 + r + 8 * hh) * S_STRIDE + (sg * 4 + si) * 16 + col] =
          acc[si][r] * scale;
  __syncthreads();

  // ---- 4. GEMM2: O = S * V via f32 WMMA 16x16x4; wave = (mt, 4 d-tiles)
  const float* Vb = Vg + (size_t)bi * Tn * Dn;   // V head-broadcast
  const int dg = wv & 3;
  v8f o[4] = {{0,0,0,0,0,0,0,0}, {0,0,0,0,0,0,0,0},
              {0,0,0,0,0,0,0,0}, {0,0,0,0,0,0,0,0}};
  const float* srow_p = Sl + (mt * 16 + col) * S_STRIDE;
  for (int sb = 0; sb < 256; sb += 4) {
    const int sr = sb + 2 * hh;
    v2f a = {srow_p[sr], srow_p[sr + 1]};
    #pragma unroll
    for (int q = 0; q < 4; ++q) {
      const int d = (dg * 4 + q) * 16 + col;
      v2f b = {Vb[sr * Dn + d], Vb[(sr + 1) * Dn + d]};
      o[q] = __builtin_amdgcn_wmma_f32_16x16x4_f32(
          false, a, false, b, (short)0, o[q], false, false);
    }
  }

  float* Ob = Og + (size_t)((bi * Hn + hi) * Tn + mbase + mt * 16) * Dn;
  #pragma unroll
  for (int q = 0; q < 4; ++q)
    #pragma unroll
    for (int r = 0; r < 8; ++r)
      Ob[(r + 8 * hh) * Dn + (dg * 4 + q) * 16 + col] = o[q][r];
}

extern "C" void kernel_launch(void* const* d_in, const int* in_sizes, int n_in,
                              void* d_out, int out_size, void* d_ws, size_t ws_size,
                              hipStream_t stream) {
  (void)in_sizes; (void)n_in; (void)out_size; (void)d_ws; (void)ws_size;
  const float* Q = (const float*)d_in[0];
  const float* K = (const float*)d_in[1];
  const float* V = (const float*)d_in[2];
  float* O = (float*)d_out;
  dim3 grid(Tn / 64, Hn, Bsz);       // (m quarter, h, b) = (4, 4, 16) = 256 blocks
  rope_attn_gfx1250<<<grid, 512, 0, stream>>>(Q, K, V, O);
}
